// Net_6451040878857
// MI455X (gfx1250) — matrix operand order repaired and checked by reading, not compiled, not run on hardware
//
#include <hip/hip_runtime.h>
#include <hip/hip_bf16.h>
#include <cmath>

// ---------------------------------------------------------------------------
// 2-layer GAT on MI455X (gfx1250, wave32).
//   L1: hp1 = x @ W1  (100000x512 @ 512x64)  -> split-bf16 WMMA
//   attention softmax over edges (incl. self loops), scatter-add messages
//   ELU -> L2: hp2 = h1 @ W2 (K=64, 16 cols) -> same WMMA kernel
//   attention again, then log_softmax(16)
// ---------------------------------------------------------------------------

typedef __attribute__((ext_vector_type(16))) __bf16 v16bf;
typedef __attribute__((ext_vector_type(8)))  float  v8f;
typedef __attribute__((ext_vector_type(4)))  float  v4f;

#define SLOPE 0.2f

__device__ __forceinline__ void atomicAddF(float* p, float v) {
    __hip_atomic_fetch_add(p, v, __ATOMIC_RELAXED, __HIP_MEMORY_SCOPE_AGENT);
}

// ---------------------------------------------------------------------------
// Split-bf16 WMMA GEMM:  Out[nRows, NCOL] = A[nRows, K] @ W[K, NCOL]
// A is fp32; decomposed per element into bf16 hi + bf16 lo on the fly.
// W is pre-split into hi/lo bf16 fragments in LDS, in this kernel's contiguous-k B-operand
// layout: frag[hilo][kt][ct][lane][j] where element j of lane L is
//   W[32*kt + 16*(L/16) + j][16*ct + (L%16)]
// Each wave owns 16 rows x all NCOL columns; x is read exactly once.
// ---------------------------------------------------------------------------
template<int K, int NCOL>
__global__ __launch_bounds__(256, 1)
void wmma_gemm_split(const float* __restrict__ A, const float* __restrict__ W,
                     float* __restrict__ Out, int nRows)
{
    constexpr int KT = K / 32;
    constexpr int CT = NCOL / 16;
    __shared__ __align__(32) __bf16 fragB[2][KT][CT][32][16];

    const int tid = threadIdx.x;

    // Cooperative pack of W into hi/lo WMMA fragments (W is tiny, L2-hot).
    const int totalSlots = KT * CT * 32;
    for (int slot = tid; slot < totalSlots; slot += 256) {
        const int lane  = slot & 31;
        const int ct    = (slot >> 5) % CT;
        const int kt    = (slot >> 5) / CT;
        const int half  = lane >> 4;
        const int col   = ct * 16 + (lane & 15);
        const int kbase = kt * 32 + half * 16;
#pragma unroll
        for (int j = 0; j < 16; ++j) {
            const float w  = W[(size_t)(kbase + j) * NCOL + col];
            const __bf16 h = (__bf16)w;
            fragB[0][kt][ct][lane][j] = h;
            fragB[1][kt][ct][lane][j] = (__bf16)(w - (float)h);
        }
    }
    __syncthreads();

    const int wave    = tid >> 5;
    const int lane    = tid & 31;
    const int rowTile = blockIdx.x * 8 + wave;
    if (rowTile * 16 >= nRows) return;          // whole-wave uniform skip
    const int rowBase = rowTile * 16;
    const int half    = lane >> 4;
    const float* __restrict__ arow = A + (size_t)(rowBase + (lane & 15)) * K;

    v8f acc[CT] = {};

    for (int kt = 0; kt < KT; ++kt) {
        // A fragment: lane holds row (lane%16), K = {k0..k0+15}
        const int k0 = kt * 32 + half * 16;
        __align__(16) float fv[16];
        *(v4f*)&fv[0]  = *(const v4f*)(arow + k0);
        *(v4f*)&fv[4]  = *(const v4f*)(arow + k0 + 4);
        *(v4f*)&fv[8]  = *(const v4f*)(arow + k0 + 8);
        *(v4f*)&fv[12] = *(const v4f*)(arow + k0 + 12);

        v16bf ahi, alo;
#pragma unroll
        for (int j = 0; j < 16; ++j) {
            const __bf16 h = (__bf16)fv[j];
            ahi[j] = h;
            alo[j] = (__bf16)(fv[j] - (float)h);
        }

#pragma unroll
        for (int t = 0; t < CT; ++t) {
            const v16bf bhi = *(const v16bf*)&fragB[0][kt][t][lane][0];
            const v16bf blo = *(const v16bf*)&fragB[1][kt][t][lane][0];
            acc[t] = __builtin_amdgcn_wmma_f32_16x16x32_bf16(
                false, ahi, false, bhi, (short)0, acc[t], false, false);
            acc[t] = __builtin_amdgcn_wmma_f32_16x16x32_bf16(
                false, ahi, false, blo, (short)0, acc[t], false, false);
            acc[t] = __builtin_amdgcn_wmma_f32_16x16x32_bf16(
                false, alo, false, bhi, (short)0, acc[t], false, false);
        }
    }

    // C/D layout: VGPR r, lane L -> row = r + 8*(L/16), col = L%16
#pragma unroll
    for (int t = 0; t < CT; ++t)
#pragma unroll
        for (int r = 0; r < 8; ++r)
            Out[(size_t)(rowBase + r + half * 8) * NCOL + t * 16 + (lane & 15)]
                = acc[t][r];
}

// al_s[n,h] = sum_c hp[n,h,c]*a_src[h,c]; al_d likewise.
template<int H, int C>
__global__ void attn_coef(const float* __restrict__ hp,
                          const float* __restrict__ a_src,
                          const float* __restrict__ a_dst,
                          float* __restrict__ al_s, float* __restrict__ al_d,
                          int n)
{
    const int idx = blockIdx.x * blockDim.x + threadIdx.x;
    if (idx >= n * H) return;
    const int node = idx / H, h = idx - node * H;
    const float* v = hp + (size_t)node * (H * C) + h * C;
    float s = 0.f, d = 0.f;
#pragma unroll
    for (int c = 0; c < C; ++c) {
        s += v[c] * a_src[h * C + c];
        d += v[c] * a_dst[h * C + c];
    }
    al_s[idx] = s;
    al_d[idx] = d;
}

// denom[d,h] += exp(leaky_relu(al_s[s,h] + al_d[d,h]))  (max-shift skipped:
// logits are O(1) here, result mathematically identical)
template<int H>
__global__ void edge_denom(const int* __restrict__ ei,
                           const float* __restrict__ al_s,
                           const float* __restrict__ al_d,
                           float* __restrict__ den, int E_, int n)
{
    const int idx = blockIdx.x * blockDim.x + threadIdx.x;
    if (idx >= (E_ + n) * H) return;
    const int e = idx / H, h = idx - e * H;
    int s, d;
    if (e < E_) { s = ei[e]; d = ei[E_ + e]; } else { s = d = e - E_; }
    float v = al_s[s * H + h] + al_d[d * H + h];
    v = v > 0.f ? v : SLOPE * v;
    atomicAddF(&den[d * H + h], __expf(v));
}

// acc[d,h,:] += hp[s,h,:] * exp(e)/denom[d,h]
template<int H, int C>
__global__ void edge_msg(const int* __restrict__ ei,
                         const float* __restrict__ al_s,
                         const float* __restrict__ al_d,
                         const float* __restrict__ den,
                         const float* __restrict__ hp,
                         float* __restrict__ acc, int E_, int n)
{
    const int idx = blockIdx.x * blockDim.x + threadIdx.x;
    if (idx >= (E_ + n) * H) return;
    const int e = idx / H, h = idx - e * H;
    int s, d;
    if (e < E_) { s = ei[e]; d = ei[E_ + e]; } else { s = d = e - E_; }
    float v = al_s[s * H + h] + al_d[d * H + h];
    v = v > 0.f ? v : SLOPE * v;
    const float alpha = __expf(v) / den[d * H + h];
    const float* hs = hp  + (size_t)s * (H * C) + h * C;
    float*       ad = acc + (size_t)d * (H * C) + h * C;
#pragma unroll
    for (int c = 0; c < C; ++c) atomicAddF(&ad[c], hs[c] * alpha);
}

__global__ void elu_bias(float* __restrict__ h, const float* __restrict__ b,
                         int total, int ncol)
{
    const int idx = blockIdx.x * blockDim.x + threadIdx.x;
    if (idx >= total) return;
    const float v = h[idx] + b[idx % ncol];
    h[idx] = v > 0.f ? v : __expf(v) - 1.f;
}

__global__ void bias_logsoftmax16(const float* __restrict__ acc,
                                  const float* __restrict__ b,
                                  float* __restrict__ out, int n)
{
    const int i = blockIdx.x * blockDim.x + threadIdx.x;
    if (i >= n) return;
    float v[16];
    float m = -1e30f;
#pragma unroll
    for (int c = 0; c < 16; ++c) {
        v[c] = acc[(size_t)i * 16 + c] + b[c];
        m = fmaxf(m, v[c]);
    }
    float s = 0.f;
#pragma unroll
    for (int c = 0; c < 16; ++c) s += __expf(v[c] - m);
    const float l = m + __logf(s);
#pragma unroll
    for (int c = 0; c < 16; ++c) out[(size_t)i * 16 + c] = v[c] - l;
}

static inline int cdiv(long long a, long long b) { return (int)((a + b - 1) / b); }

extern "C" void kernel_launch(void* const* d_in, const int* in_sizes, int n_in,
                              void* d_out, int out_size, void* d_ws, size_t ws_size,
                              hipStream_t stream)
{
    const float* x     = (const float*)d_in[0];
    const int*   ei    = (const int*)  d_in[1];
    const float* W1    = (const float*)d_in[2];
    const float* a1s   = (const float*)d_in[3];
    const float* a1d   = (const float*)d_in[4];
    const float* b1    = (const float*)d_in[5];
    const float* W2    = (const float*)d_in[6];
    const float* a2s   = (const float*)d_in[7];
    const float* a2d   = (const float*)d_in[8];
    const float* b2    = (const float*)d_in[9];
    float*       out   = (float*)d_out;

    const int N = in_sizes[0] / 512;   // 100000
    const int E = in_sizes[1] / 2;     // 1600000

    // Workspace carve-out (~75 MB total)
    char* p = (char*)d_ws;
    float* hp1   = (float*)p; p += (size_t)N * 64 * 4;
    float* al1_s = (float*)p; p += (size_t)N * 8 * 4;
    float* al1_d = (float*)p; p += (size_t)N * 8 * 4;
    float* den1  = (float*)p; p += (size_t)N * 8 * 4;
    float* acc1  = (float*)p; p += (size_t)N * 64 * 4;   // becomes h1 after ELU
    float* hp2   = (float*)p; p += (size_t)N * 16 * 4;
    float* al2_s = (float*)p; p += (size_t)N * 4;
    float* al2_d = (float*)p; p += (size_t)N * 4;
    float* den2  = (float*)p; p += (size_t)N * 4;
    float* acc2  = (float*)p; p += (size_t)N * 16 * 4;

    // Zero atomic-accumulation targets (stream-ordered; graph-capturable)
    hipMemsetAsync(den1, 0, (size_t)N * 8 * 4,  stream);
    hipMemsetAsync(acc1, 0, (size_t)N * 64 * 4, stream);
    hipMemsetAsync(den2, 0, (size_t)N * 4,      stream);
    hipMemsetAsync(acc2, 0, (size_t)N * 16 * 4, stream);

    const int gemmBlocks = cdiv(cdiv(N, 16), 8);

    // ---- Layer 1 ----
    wmma_gemm_split<512, 64><<<gemmBlocks, 256, 0, stream>>>(x, W1, hp1, N);
    attn_coef<8, 8><<<cdiv((long long)N * 8, 256), 256, 0, stream>>>(
        hp1, a1s, a1d, al1_s, al1_d, N);
    edge_denom<8><<<cdiv((long long)(E + N) * 8, 256), 256, 0, stream>>>(
        ei, al1_s, al1_d, den1, E, N);
    edge_msg<8, 8><<<cdiv((long long)(E + N) * 8, 256), 256, 0, stream>>>(
        ei, al1_s, al1_d, den1, hp1, acc1, E, N);
    elu_bias<<<cdiv((long long)N * 64, 256), 256, 0, stream>>>(
        acc1, b1, N * 64, 64);

    // ---- Layer 2 ----
    wmma_gemm_split<64, 16><<<gemmBlocks, 256, 0, stream>>>(acc1, W2, hp2, N);
    attn_coef<1, 16><<<cdiv((long long)N, 256), 256, 0, stream>>>(
        hp2, a2s, a2d, al2_s, al2_d, N);
    edge_denom<1><<<cdiv((long long)(E + N), 256), 256, 0, stream>>>(
        ei, al2_s, al2_d, den2, E, N);
    edge_msg<1, 16><<<cdiv((long long)(E + N), 256), 256, 0, stream>>>(
        ei, al2_s, al2_d, den2, hp2, acc2, E, N);
    bias_logsoftmax16<<<cdiv((long long)N, 256), 256, 0, stream>>>(
        acc2, b2, out, N);
}
